// LSTMInverter_59184649339301
// MI455X (gfx1250) — matrix operand order repaired and checked by reading, not compiled, not run on hardware
//
#include <hip/hip_runtime.h>
#include <hip/hip_bf16.h>
#include <math.h>

// ---------------------------------------------------------------------------
// LSTM language-model "inverter" for MI455X (gfx1250, wave32, WMMA).
//   B=32, E=768, V=30000, H=512, L=2, T=128
// Strategy:
//   * one-hot input GEMM -> column gather of W_ih0 (kills ~500 GFLOP of work)
//   * all recurrent/output GEMMs via v_wmma_f32_16x16x32_bf16, fp32 accum
//   * bf16 weight copies live in L2 (W_out bf16 = 30.7 MB << 192 MB L2)
//   * per-timestep dependency handled by 4 kernel launches/step (graph-safe)
// ---------------------------------------------------------------------------

typedef __bf16 bf16_t;
typedef __attribute__((ext_vector_type(16))) __bf16 v16bf;
typedef __attribute__((ext_vector_type(8)))  __bf16 v8bf;
typedef __attribute__((ext_vector_type(8)))  float  v8f;

#define B_   32
#define E_   768
#define V_   30000
#define H_   512
#define T_   128
#define G4H_ (4 * H_)   // 2048

static __device__ inline v8f zero8() {
  v8f z = {0.f, 0.f, 0.f, 0.f, 0.f, 0.f, 0.f, 0.f};
  return z;
}

static __device__ inline float sigmoidf_(float x) {
  return 1.0f / (1.0f + expf(-x));
}

// A-fragment: activation act[M][ldk] row-major bf16, tile (mtile, kbase).
// ISA 16-bit A layout: lane L holds row M=L&15; K = koff..koff+7 and 16+koff..,
// koff = (L>>4)*8.  Two aligned 16-byte loads.
static __device__ inline v16bf load_a_frag(const bf16_t* __restrict__ act,
                                           int ldk, int mtile, int kbase, int lane) {
  const int m    = mtile * 16 + (lane & 15);
  const int koff = (lane >> 4) * 8;
  const bf16_t* p = act + (size_t)m * ldk + kbase + koff;
  v8bf lo = *(const v8bf*)(p);
  v8bf hi = *(const v8bf*)(p + 16);
  v16bf a;
#pragma unroll
  for (int e = 0; e < 8; ++e) { a[e] = lo[e]; a[e + 8] = hi[e]; }
  return a;
}

// B-fragment: weights w[N][ldk] row-major bf16 (i.e. W^T GEMM), tile (ntile, kbase).
// ISA 16-bit B layout: lane L holds column n=L&15; K = koff..koff+7 and 16+koff..,
// koff = (L>>4)*8.  Two aligned 16-byte loads.
static __device__ inline v16bf load_b_frag(const bf16_t* __restrict__ w,
                                           int ldk, int ntile, int kbase, int lane) {
  const int n  = ntile * 16 + (lane & 15);
  const int k2 = (lane >> 4) * 8;
  const bf16_t* p = w + (size_t)n * ldk + kbase + k2;
  v8bf lo = *(const v8bf*)(p);
  v8bf hi = *(const v8bf*)(p + 16);
  v16bf b;
#pragma unroll
  for (int e = 0; e < 8; ++e) { b[e] = lo[e]; b[e + 8] = hi[e]; }
  return b;
}

static __device__ inline v8f wmma_bf16(v16bf a, v16bf b, v8f c) {
  // (neg_a, A, neg_b, B, c_mod, C, reuse_a, reuse_b)
  return __builtin_amdgcn_wmma_f32_16x16x32_bf16(false, a, false, b, (short)0, c,
                                                 false, false);
}

// ---------------------------------------------------------------------------
// fp32 -> bf16 weight conversion (one-time)
// ---------------------------------------------------------------------------
__global__ __launch_bounds__(256)
void lstm_convert_bf16_kernel(const float* __restrict__ src,
                              bf16_t* __restrict__ dst, int n) {
  int i = blockIdx.x * blockDim.x + threadIdx.x;
  if (i < n) dst[i] = (bf16_t)src[i];
}

// ---------------------------------------------------------------------------
// init: proj = emb @ W_proj^T + b_proj -> h0/c0 per layer; pred = -1
// 65536 outputs x 768 MACs, one-time, plain fp32.
// ---------------------------------------------------------------------------
__global__ __launch_bounds__(256)
void lstm_init_state_kernel(const float* __restrict__ emb,     // [B, E]
                            const float* __restrict__ Wproj,   // [4H, E] (=2048x768)
                            const float* __restrict__ bproj,   // [4H]
                            bf16_t* __restrict__ h0,           // [B, H] bf16
                            bf16_t* __restrict__ h1,           // [B, H] bf16
                            float* __restrict__ c0,            // [B, H]
                            float* __restrict__ c1,            // [B, H]
                            int* __restrict__ pred) {          // [B]
  int idx = blockIdx.x * blockDim.x + threadIdx.x;
  if (idx < B_) pred[idx] = -1;                 // t=0 one-hot input is all-zero
  if (idx >= B_ * G4H_) return;
  const int m = idx / G4H_;
  const int j = idx % G4H_;                     // (l, s, h) = (j/1024, (j/512)&1, j&511)
  const float* er = emb + (size_t)m * E_;
  const float* wr = Wproj + (size_t)j * E_;
  float acc = bproj[j];
  for (int k = 0; k < E_; ++k) acc = fmaf(er[k], wr[k], acc);
  const int l = j >> 10;
  const int s = (j >> 9) & 1;
  const int h = j & (H_ - 1);
  if (s == 0) {
    bf16_t* hp = (l == 0) ? h0 : h1;
    hp[m * H_ + h] = (bf16_t)acc;
  } else {
    float* cp = (l == 0) ? c0 : c1;
    cp[m * H_ + h] = acc;
  }
}

// ---------------------------------------------------------------------------
// Layer-0 LSTM cell. One wave owns one h-tile (16 h values) and computes the
// matching i/f/g/o tiles (N = h, h+512, h+1024, h+1536) so the gate fusion is
// lane-local. gates = gather(W_ih0[:,pred]) + h @ W_hh0^T + b.
// grid = 8 blocks x 128 thr (32 waves = 32 h-tiles).
// ---------------------------------------------------------------------------
__global__ __launch_bounds__(128)
void lstm_cell0_kernel(const float* __restrict__ W_ih0,   // [4H, V] fp32 (gathered)
                       const float* __restrict__ b_ih0,   // [4H]
                       const float* __restrict__ b_hh0,   // [4H]
                       const bf16_t* __restrict__ Whh0b,  // [4H, H] bf16
                       const bf16_t* __restrict__ h_in,   // [B, H] bf16
                       float* __restrict__ c_io,          // [B, H] fp32 (in-place)
                       const int* __restrict__ pred,      // [B], -1 => zero input
                       bf16_t* __restrict__ h_out) {      // [B, H] bf16
  const int lane  = threadIdx.x & 31;
  const int wid   = blockIdx.x * (blockDim.x >> 5) + (threadIdx.x >> 5);
  if (wid >= H_ / 16) return;
  const int htile = wid;

  v8f acc[4][2];
#pragma unroll
  for (int g = 0; g < 4; ++g) { acc[g][0] = zero8(); acc[g][1] = zero8(); }

  for (int kb = 0; kb < H_; kb += 32) {
    v16bf a0 = load_a_frag(h_in, H_, 0, kb, lane);
    v16bf a1 = load_a_frag(h_in, H_, 1, kb, lane);
#pragma unroll
    for (int g = 0; g < 4; ++g) {
      v16bf bf = load_b_frag(Whh0b, H_, g * (H_ / 16) + htile, kb, lane);
      acc[g][0] = wmma_bf16(a0, bf, acc[g][0]);
      acc[g][1] = wmma_bf16(a1, bf, acc[g][1]);
    }
  }

  const int nh = htile * 16 + (lane & 15);
  const int hi = lane >> 4;
  float bias[4];
#pragma unroll
  for (int g = 0; g < 4; ++g) bias[g] = b_ih0[g * H_ + nh] + b_hh0[g * H_ + nh];

#pragma unroll
  for (int mt = 0; mt < 2; ++mt) {
#pragma unroll
    for (int r = 0; r < 8; ++r) {
      const int m  = mt * 16 + hi * 8 + r;        // D layout: m = r + 8*(lane>>4)
      const int pm = pred[m];
      float gi = acc[0][mt][r] + bias[0];
      float gf = acc[1][mt][r] + bias[1];
      float gg = acc[2][mt][r] + bias[2];
      float go = acc[3][mt][r] + bias[3];
      if (pm >= 0) {                               // one-hot input -> column gather
        gi += W_ih0[((size_t)(0 * H_ + nh)) * V_ + pm];
        gf += W_ih0[((size_t)(1 * H_ + nh)) * V_ + pm];
        gg += W_ih0[((size_t)(2 * H_ + nh)) * V_ + pm];
        go += W_ih0[((size_t)(3 * H_ + nh)) * V_ + pm];
      }
      const float c_old = c_io[m * H_ + nh];
      const float cn = sigmoidf_(gf) * c_old + sigmoidf_(gi) * tanhf(gg);
      const float hn = sigmoidf_(go) * tanhf(cn);
      c_io[m * H_ + nh]  = cn;                     // element-exclusive: in-place ok
      h_out[m * H_ + nh] = (bf16_t)hn;
    }
  }
}

// ---------------------------------------------------------------------------
// Layer-1 LSTM cell: gates = x @ W_ih1^T + h @ W_hh1^T + b (x = new h0, bf16).
// ---------------------------------------------------------------------------
__global__ __launch_bounds__(128)
void lstm_cell1_kernel(const bf16_t* __restrict__ Wih1b,  // [4H, H] bf16
                       const bf16_t* __restrict__ Whh1b,  // [4H, H] bf16
                       const float* __restrict__ b_ih1,
                       const float* __restrict__ b_hh1,
                       const bf16_t* __restrict__ x_in,   // [B, H] (h0 new)
                       const bf16_t* __restrict__ h_in,   // [B, H]
                       float* __restrict__ c_io,          // [B, H]
                       bf16_t* __restrict__ h_out) {      // [B, H]
  const int lane  = threadIdx.x & 31;
  const int wid   = blockIdx.x * (blockDim.x >> 5) + (threadIdx.x >> 5);
  if (wid >= H_ / 16) return;
  const int htile = wid;

  v8f acc[4][2];
#pragma unroll
  for (int g = 0; g < 4; ++g) { acc[g][0] = zero8(); acc[g][1] = zero8(); }

  for (int kb = 0; kb < H_; kb += 32) {            // x @ W_ih1^T
    v16bf a0 = load_a_frag(x_in, H_, 0, kb, lane);
    v16bf a1 = load_a_frag(x_in, H_, 1, kb, lane);
#pragma unroll
    for (int g = 0; g < 4; ++g) {
      v16bf bf = load_b_frag(Wih1b, H_, g * (H_ / 16) + htile, kb, lane);
      acc[g][0] = wmma_bf16(a0, bf, acc[g][0]);
      acc[g][1] = wmma_bf16(a1, bf, acc[g][1]);
    }
  }
  for (int kb = 0; kb < H_; kb += 32) {            // h @ W_hh1^T
    v16bf a0 = load_a_frag(h_in, H_, 0, kb, lane);
    v16bf a1 = load_a_frag(h_in, H_, 1, kb, lane);
#pragma unroll
    for (int g = 0; g < 4; ++g) {
      v16bf bf = load_b_frag(Whh1b, H_, g * (H_ / 16) + htile, kb, lane);
      acc[g][0] = wmma_bf16(a0, bf, acc[g][0]);
      acc[g][1] = wmma_bf16(a1, bf, acc[g][1]);
    }
  }

  const int nh = htile * 16 + (lane & 15);
  const int hi = lane >> 4;
  float bias[4];
#pragma unroll
  for (int g = 0; g < 4; ++g) bias[g] = b_ih1[g * H_ + nh] + b_hh1[g * H_ + nh];

#pragma unroll
  for (int mt = 0; mt < 2; ++mt) {
#pragma unroll
    for (int r = 0; r < 8; ++r) {
      const int m = mt * 16 + hi * 8 + r;
      float gi = acc[0][mt][r] + bias[0];
      float gf = acc[1][mt][r] + bias[1];
      float gg = acc[2][mt][r] + bias[2];
      float go = acc[3][mt][r] + bias[3];
      const float c_old = c_io[m * H_ + nh];
      const float cn = sigmoidf_(gf) * c_old + sigmoidf_(gi) * tanhf(gg);
      const float hn = sigmoidf_(go) * tanhf(cn);
      c_io[m * H_ + nh]  = cn;
      h_out[m * H_ + nh] = (bf16_t)hn;
    }
  }
}

// ---------------------------------------------------------------------------
// logits[t] = h1 @ W_out^T + b_out, written straight to d_out[b][t][v].
// One wave per 16-wide V strip (1875 strips); W_out bf16 is L2-resident.
// ---------------------------------------------------------------------------
__global__ __launch_bounds__(256)
void lstm_logits_kernel(const bf16_t* __restrict__ Woutb,  // [V, H] bf16
                        const float* __restrict__ b_out,   // [V]
                        const bf16_t* __restrict__ h_in,   // [B, H] bf16
                        float* __restrict__ out,           // [B, T, V]
                        int t) {
  const int lane = threadIdx.x & 31;
  const int wid  = blockIdx.x * (blockDim.x >> 5) + (threadIdx.x >> 5);
  if (wid >= V_ / 16) return;                      // 30000/16 = 1875 exact

  v8f acc0 = zero8(), acc1 = zero8();
  for (int kb = 0; kb < H_; kb += 32) {
    v16bf a0 = load_a_frag(h_in, H_, 0, kb, lane);
    v16bf a1 = load_a_frag(h_in, H_, 1, kb, lane);
    v16bf bf = load_b_frag(Woutb, H_, wid, kb, lane);
    acc0 = wmma_bf16(a0, bf, acc0);
    acc1 = wmma_bf16(a1, bf, acc1);
  }

  const int n  = wid * 16 + (lane & 15);
  const int hi = lane >> 4;
  const float bb = b_out[n];
#pragma unroll
  for (int r = 0; r < 8; ++r) {
    const int m0 = hi * 8 + r;
    const int m1 = 16 + hi * 8 + r;
    out[((size_t)m0 * T_ + t) * V_ + n] = acc0[r] + bb;
    out[((size_t)m1 * T_ + t) * V_ + n] = acc1[r] + bb;
  }
}

// ---------------------------------------------------------------------------
// argmax over V per batch row (first-index tie-break, like jnp.argmax).
// grid = 32 blocks (one per batch), 256 threads.
// ---------------------------------------------------------------------------
__global__ __launch_bounds__(256)
void lstm_argmax_kernel(const float* __restrict__ out, int t,
                        int* __restrict__ pred) {
  const int b   = blockIdx.x;
  const int tid = threadIdx.x;
  const float* row = out + ((size_t)b * T_ + t) * V_;
  float best = -3.402823466e+38f;
  int   bidx = 0;
  for (int v = tid; v < V_; v += blockDim.x) {
    const float x = row[v];
    if (x > best) { best = x; bidx = v; }          // ascending v: first max kept
  }
  __shared__ float sv[256];
  __shared__ int   si[256];
  sv[tid] = best; si[tid] = bidx;
  __syncthreads();
  for (int s = 128; s > 0; s >>= 1) {
    if (tid < s) {
      const float ov = sv[tid + s];
      const int   oi = si[tid + s];
      if (ov > sv[tid] || (ov == sv[tid] && oi < si[tid])) { sv[tid] = ov; si[tid] = oi; }
    }
    __syncthreads();
  }
  if (tid == 0) pred[b] = si[0];
}

// ---------------------------------------------------------------------------
extern "C" void kernel_launch(void* const* d_in, const int* in_sizes, int n_in,
                              void* d_out, int out_size, void* d_ws, size_t ws_size,
                              hipStream_t stream) {
  (void)in_sizes; (void)n_in; (void)out_size; (void)ws_size;

  const float* emb   = (const float*)d_in[0];
  const float* Wproj = (const float*)d_in[1];
  const float* bproj = (const float*)d_in[2];
  const float* Wih0  = (const float*)d_in[3];
  const float* Whh0  = (const float*)d_in[4];
  const float* bih0  = (const float*)d_in[5];
  const float* bhh0  = (const float*)d_in[6];
  const float* Wih1  = (const float*)d_in[7];
  const float* Whh1  = (const float*)d_in[8];
  const float* bih1  = (const float*)d_in[9];
  const float* bhh1  = (const float*)d_in[10];
  const float* Wout  = (const float*)d_in[11];
  const float* bout  = (const float*)d_in[12];
  float* out = (float*)d_out;

  // Workspace carve-out (~37.3 MB total), 256B aligned slices.
  char*  ws  = (char*)d_ws;
  size_t off = 0;
  auto carve = [&](size_t bytes) -> void* {
    void* p = ws + off;
    off = (off + bytes + 255) & ~(size_t)255;
    return p;
  };
  const int NWHH  = G4H_ * H_;     // 2048*512
  const int NWOUT = V_ * H_;       // 30000*512
  bf16_t* Whh0b = (bf16_t*)carve((size_t)NWHH * 2);
  bf16_t* Wih1b = (bf16_t*)carve((size_t)NWHH * 2);
  bf16_t* Whh1b = (bf16_t*)carve((size_t)NWHH * 2);
  bf16_t* Woutb = (bf16_t*)carve((size_t)NWOUT * 2);
  bf16_t* h0buf[2] = { (bf16_t*)carve(B_ * H_ * 2), (bf16_t*)carve(B_ * H_ * 2) };
  bf16_t* h1buf[2] = { (bf16_t*)carve(B_ * H_ * 2), (bf16_t*)carve(B_ * H_ * 2) };
  float*  c0   = (float*)carve(B_ * H_ * 4);
  float*  c1   = (float*)carve(B_ * H_ * 4);
  int*    pred = (int*)carve(B_ * 4);

  // One-time: bf16 weight copies + initial state (deterministic each call).
  lstm_convert_bf16_kernel<<<(NWHH + 255) / 256, 256, 0, stream>>>(Whh0, Whh0b, NWHH);
  lstm_convert_bf16_kernel<<<(NWHH + 255) / 256, 256, 0, stream>>>(Wih1, Wih1b, NWHH);
  lstm_convert_bf16_kernel<<<(NWHH + 255) / 256, 256, 0, stream>>>(Whh1, Whh1b, NWHH);
  lstm_convert_bf16_kernel<<<(NWOUT + 255) / 256, 256, 0, stream>>>(Wout, Woutb, NWOUT);
  lstm_init_state_kernel<<<(B_ * G4H_) / 256, 256, 0, stream>>>(
      emb, Wproj, bproj, h0buf[0], h1buf[0], c0, c1, pred);

  // Sequential recurrence: 4 launches per step; stream order is the sync.
  const int CELL_BLOCKS   = (H_ / 16) / 4;                 // 8 (32 waves)
  const int LOGITS_BLOCKS = (V_ / 16 + 7) / 8;             // 235 (8 waves/blk)
  for (int t = 0; t < T_; ++t) {
    const int cur = t & 1, nxt = (t + 1) & 1;
    lstm_cell0_kernel<<<CELL_BLOCKS, 128, 0, stream>>>(
        Wih0, bih0, bhh0, Whh0b, h0buf[cur], c0, pred, h0buf[nxt]);
    lstm_cell1_kernel<<<CELL_BLOCKS, 128, 0, stream>>>(
        Wih1b, Whh1b, bih1, bhh1, h0buf[nxt], h1buf[cur], c1, h1buf[nxt]);
    lstm_logits_kernel<<<LOGITS_BLOCKS, 256, 0, stream>>>(
        Woutb, bout, h1buf[nxt], out, t);
    lstm_argmax_kernel<<<B_, 256, 0, stream>>>(out, t, pred);
  }
}
